// TransformerBlock_54855322305200
// MI455X (gfx1250) — matrix operand order repaired and checked by reading, not compiled, not run on hardware
//
#include <hip/hip_runtime.h>
#include <cstdint>
#include <cstddef>

// ---------------------------------------------------------------------------
// Types for CDNA5 WMMA (gfx1250, wave32)
// ---------------------------------------------------------------------------
typedef __bf16 bf16;
typedef __attribute__((ext_vector_type(16))) __bf16 v16bf;
typedef __attribute__((ext_vector_type(8)))  float  v8f;
typedef __attribute__((ext_vector_type(4)))  int    v4i;

union Frag {           // 16 bf16 = 32 bytes = one WMMA A/B operand
  uint4 q[2];
  v16bf v;
};

#define DMODEL 1024
#define NHEADS 16
#define DHEAD  64
#define DFF    2752
#define BATCH  2
#define SEQ    2048
#define BSROWS (BATCH*SEQ)

// ---------------------------------------------------------------------------
// gfx1250 async global->LDS copy (ASYNCcnt-tracked), with safe fallback
// Builtin signature (from probe diagnostics): (v4i AS1*, v4i AS3*, imm, imm)
// ---------------------------------------------------------------------------
#if __has_builtin(__builtin_amdgcn_global_load_async_to_lds_b128)
#define HAVE_ASYNC_LDS 1
#else
#define HAVE_ASYNC_LDS 0
#endif

typedef __attribute__((address_space(1))) v4i* gptr_v4i;
typedef __attribute__((address_space(3))) v4i* lptr_v4i;

__device__ __forceinline__ void async_copy_b128(const void* gsrc, void* ldst) {
#if HAVE_ASYNC_LDS
  __builtin_amdgcn_global_load_async_to_lds_b128(
      (gptr_v4i)(void*)gsrc, (lptr_v4i)ldst, 0, 0);
#else
  *(uint4*)ldst = *(const uint4*)gsrc;
#endif
}

__device__ __forceinline__ void wait_async_lds() {
#if HAVE_ASYNC_LDS
#if __has_builtin(__builtin_amdgcn_s_wait_asynccnt)
  __builtin_amdgcn_s_wait_asynccnt(0);
#else
  asm volatile("s_wait_asynccnt 0x0" ::: "memory");
#endif
#endif
}

__device__ __forceinline__ bf16 f2bf(float f) {
  union { float f; uint32_t u; } a; a.f = f;
  uint32_t u = a.u;
  u += 0x7FFFu + ((u >> 16) & 1u);          // round-to-nearest-even
  union { uint16_t s; bf16 b; } o; o.s = (uint16_t)(u >> 16);
  return o.b;
}

__device__ __forceinline__ v8f v8f_zero() {
  v8f z = {0.f, 0.f, 0.f, 0.f, 0.f, 0.f, 0.f, 0.f};
  return z;
}

// ---------------------------------------------------------------------------
// fp32 -> bf16 weight conversion
// ---------------------------------------------------------------------------
__global__ __launch_bounds__(256) void cvt_bf16_kernel(
    const float* __restrict__ in, bf16* __restrict__ out, int n) {
  int i = blockIdx.x * 256 + threadIdx.x;
  if (i < n) out[i] = f2bf(in[i]);
}

// ---------------------------------------------------------------------------
// RMSNorm: fp32 in, bf16 out (one block per row)
// ---------------------------------------------------------------------------
__global__ __launch_bounds__(256) void rmsnorm_kernel(
    const float* __restrict__ x, const float* __restrict__ g,
    bf16* __restrict__ out, int ncols) {
  const int row = blockIdx.x;
  const float* xr = x + (size_t)row * ncols;
  float ss = 0.f;
  for (int c = threadIdx.x; c < ncols; c += 256) {
    float v = xr[c];
    ss += v * v;
  }
  __shared__ float red[8];
  for (int m = 16; m >= 1; m >>= 1) ss += __shfl_xor(ss, m, 32);
  if ((threadIdx.x & 31) == 0) red[threadIdx.x >> 5] = ss;
  __syncthreads();
  if (threadIdx.x < 8) {
    float v = red[threadIdx.x];
    for (int m = 4; m >= 1; m >>= 1) v += __shfl_xor(v, m, 8);
    if (threadIdx.x == 0) red[0] = v;
  }
  __syncthreads();
  const float inv = 1.0f / sqrtf(red[0] / (float)ncols + 1e-5f);
  bf16* orow = out + (size_t)row * ncols;
  for (int c = threadIdx.x; c < ncols; c += 256)
    orow[c] = f2bf(xr[c] * inv * g[c]);
}

// ---------------------------------------------------------------------------
// BF16 WMMA GEMM: C(MxN) = A(MxK) @ B(KxN), fp32 accumulate.
// Block tile 256x64, 8 waves each computing 32x64 (2 A-frags x 4 B-frags =
// 8 WMMA per wave per K-step of 32). Double-buffered LDS; A tile staged with
// GLOBAL_LOAD_ASYNC_TO_LDS_B128 (ASYNCcnt), B staged transposed through VGPRs.
// Epilogues: 0 store bf16 | 1 store f32 | 2 +aux residual f32 | 3 SwiGLU bf16
// ---------------------------------------------------------------------------
enum { EPI_BF16 = 0, EPI_F32 = 1, EPI_RESID = 2, EPI_SWIGLU = 3 };

template <int EPI>
__global__ __launch_bounds__(256) void gemm_bf16_kernel(
    const bf16* __restrict__ A, const bf16* __restrict__ B,
    void* __restrict__ outp, const float* __restrict__ aux,
    int M, int N, int K) {
  constexpr int BM = 256, BN = 64, BK = 32;
  __shared__ __align__(16) bf16 bufA[2][BM * BK];   // [m][k] row-major, 16KB ea
  __shared__ __align__(16) bf16 bufB[2][BN * BK];   // [n][k] transposed, 4KB ea

  const int tid  = threadIdx.x;
  const int wave = tid >> 5;
  const int lane = tid & 31;
  const int lm   = lane & 15;
  const int lg   = lane >> 4;
  const int kb   = lg * 8;     // A-frag interleaved-K offset
  const int lg16 = lg * 16;    // unused (B-frag also uses kb)
  const int row0 = blockIdx.y * BM;
  const int col0 = blockIdx.x * BN;
  const int r0   = wave * 32;  // this wave's 32-row strip

  v8f acc[2][4];
#pragma unroll
  for (int a = 0; a < 2; a++)
#pragma unroll
    for (int nt = 0; nt < 4; nt++) acc[a][nt] = v8f_zero();

  // --- tile staging: A async (layout-preserving), B via VGPR transpose ---
  auto stage = [&](int k0, int buf) {
    // A: 256x32 bf16 = 1024 x b128, 4 per thread
#pragma unroll
    for (int u = 0; u < 4; u++) {
      int lin = u * 256 + tid;          // 0..1023
      int r = lin >> 2;                 // 0..255
      int c = (lin & 3) * 8;            // 0,8,16,24
      async_copy_b128(A + (size_t)(row0 + r) * K + k0 + c,
                      &bufA[buf][r * BK + c]);
    }
    // B: 32x64 bf16 loaded row-major, stored [n][k]
    {
      int kk = tid >> 3;                // 0..31
      int c0 = (tid & 7) * 8;           // 0..56
      uint4 val = *(const uint4*)(B + (size_t)(k0 + kk) * N + col0 + c0);
      const bf16* pv = (const bf16*)&val;
#pragma unroll
      for (int i = 0; i < 8; i++) bufB[buf][(c0 + i) * BK + kk] = pv[i];
    }
  };

  stage(0, 0);
  int cur = 0;
  for (int k0 = 0; k0 < K; k0 += BK) {
    wait_async_lds();                   // this wave's async copies landed
    __syncthreads();                    // everyone's tile visible
    if (k0 + BK < K) stage(k0 + BK, cur ^ 1);

    // --- compute on buf[cur] ---
    Frag a0, a1;
    a0.q[0] = *(const uint4*)(&bufA[cur][(r0 + lm) * BK + kb]);
    a0.q[1] = *(const uint4*)(&bufA[cur][(r0 + lm) * BK + 16 + kb]);
    a1.q[0] = *(const uint4*)(&bufA[cur][(r0 + 16 + lm) * BK + kb]);
    a1.q[1] = *(const uint4*)(&bufA[cur][(r0 + 16 + lm) * BK + 16 + kb]);
#pragma unroll
    for (int nt = 0; nt < 4; nt++) {
      Frag b;
      const bf16* bp = &bufB[cur][(nt * 16 + lm) * BK];
      b.q[0] = *(const uint4*)(bp + kb);
      b.q[1] = *(const uint4*)(bp + 16 + kb);
      acc[0][nt] = __builtin_amdgcn_wmma_f32_16x16x32_bf16(
          false, a0.v, false, b.v, (short)0, acc[0][nt], false, false);
      acc[1][nt] = __builtin_amdgcn_wmma_f32_16x16x32_bf16(
          false, a1.v, false, b.v, (short)0, acc[1][nt], false, false);
    }
    cur ^= 1;
  }

  // --- epilogue ---
#pragma unroll
  for (int a = 0; a < 2; a++) {
#pragma unroll
    for (int nt = 0; nt < 4; nt++) {
#pragma unroll
      for (int i = 0; i < 8; i++) {
        const int m = row0 + r0 + a * 16 + lg * 8 + i;
        const int n = col0 + nt * 16 + lm;
        const size_t idx = (size_t)m * N + n;
        const float v = acc[a][nt][i];
        if (EPI == EPI_BF16) {
          ((bf16*)outp)[idx] = f2bf(v);
        } else if (EPI == EPI_F32) {
          ((float*)outp)[idx] = v;
        } else if (EPI == EPI_RESID) {
          ((float*)outp)[idx] = v + aux[idx];
        } else {  // EPI_SWIGLU: out = silu(aux) * acc
          float x1 = aux[idx];
          float gate = x1 / (1.f + __expf(-x1));
          ((bf16*)outp)[idx] = f2bf(gate * v);
        }
      }
    }
  }
}

// ---------------------------------------------------------------------------
// Flash attention (causal), one wave per 16-query tile, dh=64.
// Q/K/V are bf16 [B, S, H*dh]. Online softmax in fp32; QK^T and P@V on WMMA.
// ---------------------------------------------------------------------------
__global__ __launch_bounds__(256) void attn_kernel(
    const bf16* __restrict__ Q, const bf16* __restrict__ Km,
    const bf16* __restrict__ V, bf16* __restrict__ O) {
  __shared__ __align__(16) bf16 ldsP[8][16 * 32];  // per-wave P staging
  const int tid  = threadIdx.x;
  const int wave = tid >> 5;
  const int lane = tid & 31;
  const int lm   = lane & 15;
  const int lg   = lane >> 4;
  const int kb   = lg * 8;
  const int lg16 = lg * 16;
  const float SCALE = 0.125f;  // 1/sqrt(64)

  const int gwave = blockIdx.x * 8 + wave;   // (b*H + h)*(S/16) + qt
  const int qt = gwave & (SEQ / 16 - 1);
  const int bh = gwave >> 7;                 // S/16 == 128
  const int h  = bh & (NHEADS - 1);
  const int b  = bh >> 4;
  const int q0 = qt * 16;

  // Q fragments (A operand): rows = queries, K-dim = dh
  Frag aq0, aq1;
  const bf16* qp = Q + ((size_t)(b * SEQ + q0 + lm)) * DMODEL + h * DHEAD;
  aq0.q[0] = *(const uint4*)(qp + kb);
  aq0.q[1] = *(const uint4*)(qp + 16 + kb);
  aq1.q[0] = *(const uint4*)(qp + 32 + kb);
  aq1.q[1] = *(const uint4*)(qp + 48 + kb);

  v8f o[4];
#pragma unroll
  for (int nt = 0; nt < 4; nt++) o[nt] = v8f_zero();
  float rm[8], rl[8];
#pragma unroll
  for (int i = 0; i < 8; i++) { rm[i] = -1e30f; rl[i] = 0.f; }

  bf16* Pw = ldsP[wave];

  for (int j = 0; j < q0 + 16; j += 32) {
    // K^T fragments (B operand): lane = key column, K-dim = dh (interleaved, as Q)
    const bf16* kpa = Km + ((size_t)(b * SEQ + j + lm)) * DMODEL + h * DHEAD;
    const bf16* kpb = Km + ((size_t)(b * SEQ + j + 16 + lm)) * DMODEL + h * DHEAD;
    Frag bkA0, bkA1, bkB0, bkB1;
    bkA0.q[0] = *(const uint4*)(kpa + kb);
    bkA0.q[1] = *(const uint4*)(kpa + 16 + kb);
    bkA1.q[0] = *(const uint4*)(kpa + 32 + kb);
    bkA1.q[1] = *(const uint4*)(kpa + 48 + kb);
    bkB0.q[0] = *(const uint4*)(kpb + kb);
    bkB0.q[1] = *(const uint4*)(kpb + 16 + kb);
    bkB1.q[0] = *(const uint4*)(kpb + 32 + kb);
    bkB1.q[1] = *(const uint4*)(kpb + 48 + kb);

    v8f s0 = v8f_zero(), s1 = v8f_zero();
    s0 = __builtin_amdgcn_wmma_f32_16x16x32_bf16(false, aq0.v, false, bkA0.v,
                                                 (short)0, s0, false, false);
    s0 = __builtin_amdgcn_wmma_f32_16x16x32_bf16(false, aq1.v, false, bkA1.v,
                                                 (short)0, s0, false, false);
    s1 = __builtin_amdgcn_wmma_f32_16x16x32_bf16(false, aq0.v, false, bkB0.v,
                                                 (short)0, s1, false, false);
    s1 = __builtin_amdgcn_wmma_f32_16x16x32_bf16(false, aq1.v, false, bkB1.v,
                                                 (short)0, s1, false, false);

    // online softmax per query row (row = lg*8 + i, reduce across 16 lanes)
#pragma unroll
    for (int i = 0; i < 8; i++) {
      const int qrow = q0 + lg * 8 + i;
      const int k0i = j + lm;
      const int k1i = j + 16 + lm;
      float x0 = (k0i <= qrow) ? s0[i] * SCALE : -1e30f;
      float x1 = (k1i <= qrow) ? s1[i] * SCALE : -1e30f;
      float t = fmaxf(x0, x1);
      for (int m = 8; m >= 1; m >>= 1) t = fmaxf(t, __shfl_xor(t, m, 16));
      float mnew = fmaxf(rm[i], t);
      float p0 = __expf(x0 - mnew);
      float p1 = __expf(x1 - mnew);
      float ps = p0 + p1;
      for (int m = 8; m >= 1; m >>= 1) ps += __shfl_xor(ps, m, 16);
      float sc = __expf(rm[i] - mnew);
      rl[i] = rl[i] * sc + ps;
      rm[i] = mnew;
#pragma unroll
      for (int nt = 0; nt < 4; nt++) o[nt][i] *= sc;
      // C-layout -> A-layout reshape via LDS
      Pw[(lg * 8 + i) * 32 + lm]      = f2bf(p0);
      Pw[(lg * 8 + i) * 32 + 16 + lm] = f2bf(p1);
    }
    asm volatile("s_wait_dscnt 0x0" ::: "memory");

    // P fragment (A operand, 16x32)
    Frag ap;
    ap.q[0] = *(const uint4*)(Pw + lm * 32 + kb);
    ap.q[1] = *(const uint4*)(Pw + lm * 32 + 16 + kb);

    // V fragments (B operand: K-dim = key rows, lane = dh column) + P@V
#pragma unroll
    for (int nt = 0; nt < 4; nt++) {
      Frag bv;
      bf16* e = (bf16*)&bv;
#pragma unroll
      for (int ee = 0; ee < 16; ee++) {
        int key = j + kb + ee + (ee >= 8 ? 8 : 0);
        e[ee] = V[((size_t)(b * SEQ + key)) * DMODEL + h * DHEAD + nt * 16 + lm];
      }
      o[nt] = __builtin_amdgcn_wmma_f32_16x16x32_bf16(
          false, ap.v, false, bv.v, (short)0, o[nt], false, false);
    }
  }

  // finalize: divide by softmax denom, store bf16 [B,S,H*dh]
#pragma unroll
  for (int i = 0; i < 8; i++) {
    const float inv = 1.0f / rl[i];
    const int qrow = q0 + lg * 8 + i;
#pragma unroll
    for (int nt = 0; nt < 4; nt++) {
      O[((size_t)(b * SEQ + qrow)) * DMODEL + h * DHEAD + nt * 16 + lm] =
          f2bf(o[nt][i] * inv);
    }
  }
}

// ---------------------------------------------------------------------------
// Orchestration
// ---------------------------------------------------------------------------
extern "C" void kernel_launch(void* const* d_in, const int* in_sizes, int n_in,
                              void* d_out, int out_size, void* d_ws, size_t ws_size,
                              hipStream_t stream) {
  (void)in_sizes; (void)n_in; (void)out_size; (void)ws_size;
  const float* x  = (const float*)d_in[0];
  const float* g1 = (const float*)d_in[1];
  const float* g2 = (const float*)d_in[2];
  const float* wq = (const float*)d_in[3];
  const float* wk = (const float*)d_in[4];
  const float* wv = (const float*)d_in[5];
  const float* wo = (const float*)d_in[6];
  const float* w1 = (const float*)d_in[7];
  const float* w2 = (const float*)d_in[8];
  const float* w3 = (const float*)d_in[9];
  float* out = (float*)d_out;

  char* ws = (char*)d_ws;
  size_t off = 0;
  auto alloc = [&](size_t bytes) -> char* {
    char* p = ws + off;
    off += (bytes + 255) & ~(size_t)255;
    return p;
  };
  const size_t NW  = (size_t)DMODEL * DMODEL;
  const size_t NWF = (size_t)DMODEL * DFF;
  const size_t NX  = (size_t)BSROWS * DMODEL;
  const size_t NXF = (size_t)BSROWS * DFF;

  bf16* wq_b   = (bf16*)alloc(NW * 2);
  bf16* wk_b   = (bf16*)alloc(NW * 2);
  bf16* wv_b   = (bf16*)alloc(NW * 2);
  bf16* wo_b   = (bf16*)alloc(NW * 2);
  bf16* w1_b   = (bf16*)alloc(NWF * 2);
  bf16* w3_b   = (bf16*)alloc(NWF * 2);
  bf16* w2_b   = (bf16*)alloc(NWF * 2);
  bf16* xn_b   = (bf16*)alloc(NX * 2);
  bf16* q_b    = (bf16*)alloc(NX * 2);
  bf16* k_b    = (bf16*)alloc(NX * 2);
  bf16* v_b    = (bf16*)alloc(NX * 2);
  bf16* attn_b = (bf16*)alloc(NX * 2);
  bf16* hn_b   = (bf16*)alloc(NX * 2);
  bf16* gate_b = (bf16*)alloc(NXF * 2);
  float* hres  = (float*)alloc(NX * 4);
  float* x1    = (float*)alloc(NXF * 4);

  auto cvt = [&](const float* src, bf16* dst, size_t n) {
    cvt_bf16_kernel<<<dim3((unsigned)((n + 255) / 256)), dim3(256), 0, stream>>>(
        src, dst, (int)n);
  };
  cvt(wq, wq_b, NW);
  cvt(wk, wk_b, NW);
  cvt(wv, wv_b, NW);
  cvt(wo, wo_b, NW);
  cvt(w1, w1_b, NWF);
  cvt(w3, w3_b, NWF);
  cvt(w2, w2_b, NWF);

  // norm1
  rmsnorm_kernel<<<dim3(BSROWS), dim3(256), 0, stream>>>(x, g1, xn_b, DMODEL);

  // QKV projections
  const dim3 gD(DMODEL / 64, BSROWS / 256), gF(DFF / 64, BSROWS / 256), blk(256);
  gemm_bf16_kernel<EPI_BF16><<<gD, blk, 0, stream>>>(xn_b, wq_b, q_b, nullptr,
                                                     BSROWS, DMODEL, DMODEL);
  gemm_bf16_kernel<EPI_BF16><<<gD, blk, 0, stream>>>(xn_b, wk_b, k_b, nullptr,
                                                     BSROWS, DMODEL, DMODEL);
  gemm_bf16_kernel<EPI_BF16><<<gD, blk, 0, stream>>>(xn_b, wv_b, v_b, nullptr,
                                                     BSROWS, DMODEL, DMODEL);

  // causal flash attention: B*H*(S/16) waves, 8 waves per block
  attn_kernel<<<dim3(BATCH * NHEADS * (SEQ / 16) / 8), blk, 0, stream>>>(
      q_b, k_b, v_b, attn_b);

  // output projection + residual
  gemm_bf16_kernel<EPI_RESID><<<gD, blk, 0, stream>>>(attn_b, wo_b, hres, x,
                                                      BSROWS, DMODEL, DMODEL);

  // norm2
  rmsnorm_kernel<<<dim3(BSROWS), dim3(256), 0, stream>>>(hres, g2, hn_b, DMODEL);

  // SwiGLU FFN
  gemm_bf16_kernel<EPI_F32><<<gF, blk, 0, stream>>>(hn_b, w1_b, x1, nullptr,
                                                    BSROWS, DFF, DMODEL);
  gemm_bf16_kernel<EPI_SWIGLU><<<gF, blk, 0, stream>>>(hn_b, w3_b, gate_b, x1,
                                                       BSROWS, DFF, DMODEL);
  gemm_bf16_kernel<EPI_RESID><<<gD, blk, 0, stream>>>(gate_b, w2_b, out, hres,
                                                      BSROWS, DMODEL, DFF);
}
